// TorchMHC_50714973831641
// MI455X (gfx1250) — matrix operand order repaired and checked by reading, not compiled, not run on hardware
//
#include <hip/hip_runtime.h>
#include <hip/hip_bf16.h>

#define HC_ 4
#define C_ 2048
#define TOK_ (HC_ * C_)   // 8192 elements per token
#define BK_ 32
#define BM_ 128
#define BN_ 64

typedef __attribute__((ext_vector_type(8)))  float   v8f;
typedef __attribute__((ext_vector_type(16))) __bf16  v16bf;

union Frag { uint4 u[2]; v16bf v; };

#if defined(__gfx1250__) && __has_builtin(__builtin_amdgcn_global_load_async_to_lds_b128)
#define HAVE_ASYNC_LDS 1
#else
#define HAVE_ASYNC_LDS 0
#endif

#if HAVE_ASYNC_LDS
typedef int v4i_gcc __attribute__((vector_size(16)));
typedef __attribute__((address_space(1))) v4i_gcc* gp1_t;
typedef __attribute__((address_space(3))) v4i_gcc* lp3_t;
#define ASYNC_CP_B128(gsrc, ldst) \
    __builtin_amdgcn_global_load_async_to_lds_b128((gp1_t)(gsrc), (lp3_t)(ldst), 0, 0)
__device__ __forceinline__ void async_wait0() {
#if __has_builtin(__builtin_amdgcn_s_wait_asynccnt)
    __builtin_amdgcn_s_wait_asynccnt(0);
#else
    asm volatile("s_wait_asynccnt 0x0" ::: "memory");
#endif
}
#endif

// ---------- bf16 helpers (bit-exact RNE, no reliance on __bf16 arithmetic) ----------
__device__ __forceinline__ unsigned short f2bf(float f) {
    union { float f; unsigned u; } a; a.f = f;
    unsigned u = a.u;
    u += 0x7FFFu + ((u >> 16) & 1u);          // round to nearest even
    return (unsigned short)(u >> 16);
}
__device__ __forceinline__ float bf2f(unsigned short h) {
    union { unsigned u; float f; } a; a.u = ((unsigned)h) << 16;
    return a.f;
}

// ---------------------------------------------------------------------------
// Kernel 1: transpose W (f32 [K][N]) into bf16 hi/lo planes stored [N][K].
// LDS-tiled so both the global read and the global write are coalesced.
// One-off cost (16.8 MB read, 16.8 MB written, then L2-resident).
// ---------------------------------------------------------------------------
__global__ __launch_bounds__(256) void mhc_prep_wt(const float* __restrict__ W,
                                                   unsigned short* __restrict__ Wthi,
                                                   unsigned short* __restrict__ Wtlo) {
    __shared__ float st[64][65];                   // +1 pad: conflict-free transpose
    const int k0 = blockIdx.x * 64;
    const int n0 = blockIdx.y * 64;
    const int tr = threadIdx.x >> 4;               // 0..15
    const int tc = (threadIdx.x & 15) * 4;         // 0..60

#pragma unroll
    for (int rr = 0; rr < 64; rr += 16) {
        float4 w = *(const float4*)(W + (size_t)(k0 + tr + rr) * C_ + n0 + tc);
        st[tr + rr][tc + 0] = w.x;
        st[tr + rr][tc + 1] = w.y;
        st[tr + rr][tc + 2] = w.z;
        st[tr + rr][tc + 3] = w.w;
    }
    __syncthreads();

#pragma unroll
    for (int rr = 0; rr < 64; rr += 16) {
        const int n = tr + rr;                     // local n row of Wt
        union { unsigned short h[4]; unsigned long long u; } ph, pl;
#pragma unroll
        for (int j = 0; j < 4; ++j) {
            float v = st[tc + j][n];
            ph.h[j] = f2bf(v);
            pl.h[j] = f2bf(v - bf2f(ph.h[j]));
        }
        *(unsigned long long*)(Wthi + (size_t)(n0 + n) * C_ + k0 + tc) = ph.u;
        *(unsigned long long*)(Wtlo + (size_t)(n0 + n) * C_ + k0 + tc) = pl.u;
    }
}

// ---------------------------------------------------------------------------
// Kernel 2: per-token coefficients (RMS norm -> phi projection -> sigmoids +
// 8-iteration Sinkhorn) and x_in = sum_i h_pre[i] * x[i,:] as bf16 hi/lo.
// One block (8 wave32 waves) per token; token staged in LDS.
// ---------------------------------------------------------------------------
__global__ __launch_bounds__(256) void mhc_coeff_xin(
    const float* __restrict__ x,        // [M][HC][C]
    const float* __restrict__ phi,      // [8192][24]
    const float* __restrict__ bias,     // [24]
    const float* __restrict__ a_pre_p,
    const float* __restrict__ a_post_p,
    const float* __restrict__ a_res_p,
    float* __restrict__ coeff,          // [M][24]  (0..3 pre, 4..7 post, 8..23 res)
    unsigned short* __restrict__ xin_hi,// [M][C]
    unsigned short* __restrict__ xin_lo)
{
    __shared__ float sx[TOK_];          // 32 KB token stage
    __shared__ float swred[8];
    __shared__ float slog[8][24];
    __shared__ float slogit[24];
    __shared__ float srms;
    __shared__ float scoef[24];

    const int t    = blockIdx.x;
    const int tid  = threadIdx.x;
    const int lane = tid & 31;
    const int wv   = tid >> 5;
    const float* xt = x + (size_t)t * TOK_;

    // pass 1: stage token into LDS, accumulate sum of squares
    float ss = 0.f;
#pragma unroll
    for (int j = 0; j < TOK_ / 256; ++j) {
        int k = tid + 256 * j;
        float v = xt[k];
        sx[k] = v;
        ss += v * v;
    }
    ss += __shfl_xor(ss, 16, 32);
    ss += __shfl_xor(ss,  8, 32);
    ss += __shfl_xor(ss,  4, 32);
    ss += __shfl_xor(ss,  2, 32);
    ss += __shfl_xor(ss,  1, 32);
    if (lane == 0) swred[wv] = ss;

    // pass 2: partial logits (each thread only touches the sx entries it wrote)
    float pl[24];
#pragma unroll
    for (int q = 0; q < 24; ++q) pl[q] = 0.f;
    for (int j = 0; j < TOK_ / 256; ++j) {
        int k = tid + 256 * j;
        float v = sx[k];
        const float* pr = phi + (size_t)k * 24;
#pragma unroll
        for (int q = 0; q < 24; ++q) pl[q] += v * pr[q];
    }
#pragma unroll
    for (int q = 0; q < 24; ++q) {
        pl[q] += __shfl_xor(pl[q], 16, 32);
        pl[q] += __shfl_xor(pl[q],  8, 32);
        pl[q] += __shfl_xor(pl[q],  4, 32);
        pl[q] += __shfl_xor(pl[q],  2, 32);
        pl[q] += __shfl_xor(pl[q],  1, 32);
    }
    if (lane == 0) {
#pragma unroll
        for (int q = 0; q < 24; ++q) slog[wv][q] = pl[q];
    }
    __syncthreads();

    if (tid < 24) {
        float s = 0.f;
        for (int w = 0; w < 8; ++w) s += slog[w][tid];
        slogit[tid] = s;
    }
    if (tid == 0) {
        float s = 0.f;
        for (int w = 0; w < 8; ++w) s += swred[w];
        srms = rsqrtf(s * (1.0f / TOK_) + 1e-6f);
    }
    __syncthreads();

    if (tid == 0) {
        const float rinv  = srms;
        const float apre  = a_pre_p[0];
        const float apost = a_post_p[0];
        const float ares  = a_res_p[0];
        float lg[24];
#pragma unroll
        for (int q = 0; q < 24; ++q) lg[q] = slogit[q] * rinv + bias[q];

        float hp[4], ho[4], Mm[16];
#pragma unroll
        for (int i = 0; i < 4; ++i) hp[i] = 1.f / (1.f + expf(-apre  * lg[i]))     + 1e-4f;
#pragma unroll
        for (int i = 0; i < 4; ++i) ho[i] = 2.f / (1.f + expf(-apost * lg[4 + i]));
#pragma unroll
        for (int i = 0; i < 16; ++i) Mm[i] = expf(ares * lg[8 + i]);

        for (int it = 0; it < 8; ++it) {
            for (int o = 0; o < 4; ++o) {
                float rs = Mm[o*4] + Mm[o*4+1] + Mm[o*4+2] + Mm[o*4+3] + 1e-6f;
                for (int i = 0; i < 4; ++i) Mm[o*4+i] /= rs;
            }
            for (int i = 0; i < 4; ++i) {
                float cs = Mm[i] + Mm[4+i] + Mm[8+i] + Mm[12+i] + 1e-6f;
                for (int o = 0; o < 4; ++o) Mm[o*4+i] /= cs;
            }
        }
        float* cf = coeff + (size_t)t * 24;
#pragma unroll
        for (int i = 0; i < 4;  ++i) { scoef[i]   = hp[i]; cf[i]   = hp[i]; }
#pragma unroll
        for (int i = 0; i < 4;  ++i) { scoef[4+i] = ho[i]; cf[4+i] = ho[i]; }
#pragma unroll
        for (int i = 0; i < 16; ++i) { scoef[8+i] = Mm[i]; cf[8+i] = Mm[i]; }
    }
    __syncthreads();

    // x_in -> bf16 hi/lo
    const float h0 = scoef[0], h1 = scoef[1], h2 = scoef[2], h3 = scoef[3];
#pragma unroll
    for (int j = 0; j < C_ / 256; ++j) {
        int c = tid + 256 * j;
        float v = h0 * sx[c] + h1 * sx[C_ + c] + h2 * sx[2*C_ + c] + h3 * sx[3*C_ + c];
        unsigned short hi = f2bf(v);
        unsigned short lo = f2bf(v - bf2f(hi));
        xin_hi[(size_t)t * C_ + c] = hi;
        xin_lo[(size_t)t * C_ + c] = lo;
    }
}

// ---------------------------------------------------------------------------
// Kernel 3: F = Xin @ W via bf16 WMMA with hi/lo error compensation, fused
// epilogue  out[m,o,n] = sum_i h_res[m][o][i]*x[m,i,n] + h_post[m][o]*(F+Wb).
// Block tile 128x64, 8 wave32 waves, wave tile 32x32 (four 16x16 f32 accs).
// LDS double-buffered, one barrier per k-chunk. Staging uses gfx1250 async
// global->LDS copies (ASYNCcnt) when available, bypassing VGPRs entirely;
// otherwise falls back to global_load_b128 -> ds_store_b128.
// ---------------------------------------------------------------------------
__global__ __launch_bounds__(256) void mhc_gemm(
    const unsigned short* __restrict__ Ahi, const unsigned short* __restrict__ Alo, // [M][C]
    const unsigned short* __restrict__ Whi, const unsigned short* __restrict__ Wlo, // [N][K] transposed
    const float* __restrict__ coeff,  // [M][24]
    const float* __restrict__ x,      // [M][HC][C]
    const float* __restrict__ Wb,     // [C]
    float* __restrict__ out)          // [M][HC][C]
{
    __shared__ __attribute__((aligned(16))) unsigned short sAh[2][BM_][BK_]; // 16 KB
    __shared__ __attribute__((aligned(16))) unsigned short sAl[2][BM_][BK_]; // 16 KB
    __shared__ __attribute__((aligned(16))) unsigned short sBh[2][BN_][BK_]; //  8 KB ([n][k])
    __shared__ __attribute__((aligned(16))) unsigned short sBl[2][BN_][BK_]; //  8 KB

    const int tid  = threadIdx.x;
    const int lane = tid & 31;
    const int wv   = tid >> 5;
    const int half = lane >> 4;
    const int l16  = lane & 15;
    const int m0   = blockIdx.x * BM_;
    const int n0   = blockIdx.y * BN_;
    const int wm   = wv & 3;       // 4 waves along M, 32 rows each
    const int wn   = wv >> 2;      // 2 waves along N, 32 cols each

    v8f acc00 = {}, acc01 = {}, acc10 = {}, acc11 = {};

    // staging indices: A 128x32 (16 bf16/thread/plane), B 64x32 (8 bf16/thread/plane)
    const int ar = tid >> 2, ac = (tid & 3) * 8;   // A rows ar, ar+64
    const int br = tid >> 2, bc = (tid & 3) * 8;   // B row (n), col (k)

    const unsigned short* pAh = Ahi + (size_t)(m0 + ar) * C_ + ac;
    const unsigned short* pAl = Alo + (size_t)(m0 + ar) * C_ + ac;
    const unsigned short* pBh = Whi + (size_t)(n0 + br) * C_ + bc;
    const unsigned short* pBl = Wlo + (size_t)(n0 + br) * C_ + bc;

    // prologue: stage chunk 0 into buffer 0
#if HAVE_ASYNC_LDS
    {
        ASYNC_CP_B128(pAh,                   &sAh[0][ar][ac]);
        ASYNC_CP_B128(pAh + (size_t)64 * C_, &sAh[0][ar + 64][ac]);
        ASYNC_CP_B128(pAl,                   &sAl[0][ar][ac]);
        ASYNC_CP_B128(pAl + (size_t)64 * C_, &sAl[0][ar + 64][ac]);
        ASYNC_CP_B128(pBh,                   &sBh[0][br][bc]);
        ASYNC_CP_B128(pBl,                   &sBl[0][br][bc]);
        async_wait0();
    }
#else
    {
        uint4 a0 = *(const uint4*)(pAh);
        uint4 a1 = *(const uint4*)(pAh + (size_t)64 * C_);
        uint4 a2 = *(const uint4*)(pAl);
        uint4 a3 = *(const uint4*)(pAl + (size_t)64 * C_);
        uint4 b0 = *(const uint4*)(pBh);
        uint4 b1 = *(const uint4*)(pBl);
        *(uint4*)&sAh[0][ar][ac]      = a0;
        *(uint4*)&sAh[0][ar + 64][ac] = a1;
        *(uint4*)&sAl[0][ar][ac]      = a2;
        *(uint4*)&sAl[0][ar + 64][ac] = a3;
        *(uint4*)&sBh[0][br][bc]      = b0;
        *(uint4*)&sBl[0][br][bc]      = b1;
    }
#endif
    __syncthreads();

    int p = 0;
    for (int k0 = 0; k0 < C_; k0 += BK_) {
        const bool has_next = (k0 + BK_) < C_;
#if HAVE_ASYNC_LDS
        if (has_next) {
            const int kn = k0 + BK_;
            const int q = p ^ 1;
            ASYNC_CP_B128(pAh + kn,                   &sAh[q][ar][ac]);
            ASYNC_CP_B128(pAh + (size_t)64 * C_ + kn, &sAh[q][ar + 64][ac]);
            ASYNC_CP_B128(pAl + kn,                   &sAl[q][ar][ac]);
            ASYNC_CP_B128(pAl + (size_t)64 * C_ + kn, &sAl[q][ar + 64][ac]);
            ASYNC_CP_B128(pBh + kn,                   &sBh[q][br][bc]);
            ASYNC_CP_B128(pBl + kn,                   &sBl[q][br][bc]);
        }
#else
        uint4 a0, a1, a2, a3, b0, b1;
        if (has_next) {
            const int kn = k0 + BK_;
            a0 = *(const uint4*)(pAh + kn);
            a1 = *(const uint4*)(pAh + (size_t)64 * C_ + kn);
            a2 = *(const uint4*)(pAl + kn);
            a3 = *(const uint4*)(pAl + (size_t)64 * C_ + kn);
            b0 = *(const uint4*)(pBh + kn);
            b1 = *(const uint4*)(pBl + kn);
        }
#endif

        // fragments (ISA VGPR striping => two contiguous b128 LDS loads each)
        Frag ah0, ah1, al0, al1, bh0, bh1, bl0, bl1;
        const int rA0 = wm * 32 + l16;
        const int rA1 = rA0 + 16;
        ah0.u[0] = *(const uint4*)&sAh[p][rA0][half * 8];
        ah0.u[1] = *(const uint4*)&sAh[p][rA0][16 + half * 8];
        ah1.u[0] = *(const uint4*)&sAh[p][rA1][half * 8];
        ah1.u[1] = *(const uint4*)&sAh[p][rA1][16 + half * 8];
        al0.u[0] = *(const uint4*)&sAl[p][rA0][half * 8];
        al0.u[1] = *(const uint4*)&sAl[p][rA0][16 + half * 8];
        al1.u[0] = *(const uint4*)&sAl[p][rA1][half * 8];
        al1.u[1] = *(const uint4*)&sAl[p][rA1][16 + half * 8];
        const int cB0 = wn * 32 + l16;
        const int cB1 = cB0 + 16;
        bh0.u[0] = *(const uint4*)&sBh[p][cB0][half * 8];
        bh0.u[1] = *(const uint4*)&sBh[p][cB0][16 + half * 8];
        bh1.u[0] = *(const uint4*)&sBh[p][cB1][half * 8];
        bh1.u[1] = *(const uint4*)&sBh[p][cB1][16 + half * 8];
        bl0.u[0] = *(const uint4*)&sBl[p][cB0][half * 8];
        bl0.u[1] = *(const uint4*)&sBl[p][cB0][16 + half * 8];
        bl1.u[0] = *(const uint4*)&sBl[p][cB1][half * 8];
        bl1.u[1] = *(const uint4*)&sBl[p][cB1][16 + half * 8];

        // hi/lo compensated accumulation: A*W ~= Ah*Wh + Al*Wh + Ah*Wl
        acc00 = __builtin_amdgcn_wmma_f32_16x16x32_bf16(false, ah0.v, false, bh0.v, (short)0, acc00, false, false);
        acc01 = __builtin_amdgcn_wmma_f32_16x16x32_bf16(false, ah0.v, false, bh1.v, (short)0, acc01, false, false);
        acc10 = __builtin_amdgcn_wmma_f32_16x16x32_bf16(false, ah1.v, false, bh0.v, (short)0, acc10, false, false);
        acc11 = __builtin_amdgcn_wmma_f32_16x16x32_bf16(false, ah1.v, false, bh1.v, (short)0, acc11, false, false);
        acc00 = __builtin_amdgcn_wmma_f32_16x16x32_bf16(false, al0.v, false, bh0.v, (short)0, acc00, false, false);
        acc01 = __builtin_amdgcn_wmma_f32_16x16x32_bf16(false, al0.v, false, bh1.v, (short)0, acc01, false, false);
        acc10 = __builtin_amdgcn_wmma_f32_16x16x32_bf16(false, al1.v, false, bh0.v, (short)0, acc10, false, false);
        acc11 = __builtin_amdgcn_wmma_f32_16x16x32_bf16(false, al1.v, false, bh1.v, (short)0, acc11, false, false);
        acc00 = __builtin_amdgcn_wmma_f32_16x16x32_bf16(false, ah0.v, false, bl0.v, (short)0, acc00, false, false);
        acc01 = __builtin_amdgcn_wmma_f32_16x16x32_bf16(false, ah0.v, false, bl1.v, (short)0, acc01, false, false);
        acc10 = __builtin_amdgcn_wmma_f32_16x16x32_bf16(false, ah1.v, false, bl0.v, (short)0, acc10, false, false);
        acc11 = __builtin_amdgcn_wmma_f32_16x16x32_bf16(false, ah1.v, false, bl1.v, (short)0, acc11, false, false);

#if HAVE_ASYNC_LDS
        if (has_next) async_wait0();
#else
        if (has_next) {
            const int q = p ^ 1;
            *(uint4*)&sAh[q][ar][ac]      = a0;
            *(uint4*)&sAh[q][ar + 64][ac] = a1;
            *(uint4*)&sAl[q][ar][ac]      = a2;
            *(uint4*)&sAl[q][ar + 64][ac] = a3;
            *(uint4*)&sBh[q][br][bc]      = b0;
            *(uint4*)&sBl[q][br][bc]      = b1;
        }
#endif
        __syncthreads();
        p ^= 1;
    }

    // fused epilogue; C/D layout: VGPR r -> (M = r + half*8, N = l16)
#pragma unroll
    for (int mt = 0; mt < 2; ++mt) {
#pragma unroll
        for (int nt = 0; nt < 2; ++nt) {
            v8f acc = (mt == 0) ? ((nt == 0) ? acc00 : acc01)
                                : ((nt == 0) ? acc10 : acc11);
            const int n = n0 + wn * 32 + nt * 16 + l16;
            const float wbn = Wb[n];
#pragma unroll
            for (int r = 0; r < 8; ++r) {
                const int m = m0 + wm * 32 + mt * 16 + r + half * 8;
                const float f = acc[r] + wbn;
                const float* cf = coeff + (size_t)m * 24;      // lane-uniform (broadcast)
                const float* xm = x + (size_t)m * TOK_ + n;
                const float x0 = xm[0], x1 = xm[C_], x2 = xm[2*C_], x3 = xm[3*C_];
                float* om = out + (size_t)m * TOK_ + n;
#pragma unroll
                for (int o = 0; o < 4; ++o) {
                    const float* hr = cf + 8 + o * 4;
                    om[(size_t)o * C_] = cf[4 + o] * f
                                       + hr[0] * x0 + hr[1] * x1 + hr[2] * x2 + hr[3] * x3;
                }
            }
        }
    }
}

// ---------------------------------------------------------------------------
extern "C" void kernel_launch(void* const* d_in, const int* in_sizes, int n_in,
                              void* d_out, int out_size, void* d_ws, size_t ws_size,
                              hipStream_t stream) {
    const float* x      = (const float*)d_in[0];
    const float* phi    = (const float*)d_in[1];
    const float* bias   = (const float*)d_in[2];
    const float* a_pre  = (const float*)d_in[3];
    const float* a_post = (const float*)d_in[4];
    const float* a_res  = (const float*)d_in[5];
    const float* W      = (const float*)d_in[6];
    const float* Wb     = (const float*)d_in[7];
    float* out = (float*)d_out;

    const int Mtot = in_sizes[0] / TOK_;     // tokens (4096)

    // workspace layout (~48.5 MB total)
    char* ws = (char*)d_ws;
    float* coeff = (float*)ws;
    size_t off = ((size_t)Mtot * 24 * sizeof(float) + 255) & ~(size_t)255;
    unsigned short* Ahi  = (unsigned short*)(ws + off); off += (size_t)Mtot * C_ * 2;
    unsigned short* Alo  = (unsigned short*)(ws + off); off += (size_t)Mtot * C_ * 2;
    unsigned short* Wthi = (unsigned short*)(ws + off); off += (size_t)C_ * C_ * 2;
    unsigned short* Wtlo = (unsigned short*)(ws + off); off += (size_t)C_ * C_ * 2;

    dim3 tgrid(C_ / 64, C_ / 64);
    mhc_prep_wt<<<tgrid, 256, 0, stream>>>(W, Wthi, Wtlo);
    mhc_coeff_xin<<<Mtot, 256, 0, stream>>>(x, phi, bias, a_pre, a_post, a_res,
                                            coeff, Ahi, Alo);
    dim3 grid(Mtot / BM_, C_ / BN_);
    mhc_gemm<<<grid, 256, 0, stream>>>(Ahi, Alo, Wthi, Wtlo, coeff, x, Wb, out);
}
